// GATv2_45277545234536
// MI455X (gfx1250) — matrix operand order repaired and checked by reading, not compiled, not run on hardware
//
#include <hip/hip_runtime.h>

#define NNODE  50000
#define NFEAT  256
#define NHID   64
#define NCLASS 40
#define NHEADS 8
#define NEDGE  800000
#define NH     (NHEADS * NHID)   // 512
#define ALPHA  0.2f

typedef __attribute__((ext_vector_type(16))) __bf16 bhalf16;
typedef __attribute__((ext_vector_type(8)))  float  floatx8;

// ---------- helpers ----------
__device__ inline unsigned short f2bf(float f) {
    unsigned u = __float_as_uint(f);
    unsigned r = u + 0x7FFFu + ((u >> 16) & 1u);   // RNE
    return (unsigned short)(r >> 16);
}
__device__ inline unsigned fenc(float f) {        // order-preserving float->uint
    unsigned u = __float_as_uint(f);
    return (u & 0x80000000u) ? ~u : (u | 0x80000000u);
}
__device__ inline float fdec(unsigned k) {
    unsigned u = (k & 0x80000000u) ? (k & 0x7FFFFFFFu) : ~k;
    return __uint_as_float(u);
}
// A fragment: 16x32 bf16, lane = (m = lane&15, kh = lane>>4)
// lane holds K = kh*16 .. kh*16+15 contiguously
__device__ inline bhalf16 load_a_frag_p(const unsigned short* __restrict__ rowbase,
                                        int k0, int kh) {
    union { bhalf16 v; uint4 q[2]; } u;
    const unsigned short* p = rowbase + k0 + kh * 16;
    u.q[0] = *(const uint4*)(p);
    u.q[1] = *(const uint4*)(p + 8);
    return u.v;
}
// B fragment: 32x16 bf16 from pre-transposed B^T [N x K], lane = (n = lane&15, kh = lane>>4)
// lane holds K = kh*16 .. kh*16+15 contiguously -> single 32B load
__device__ inline bhalf16 load_b_frag_p(const unsigned short* __restrict__ rowbase,
                                        int k0, int kh) {
    union { bhalf16 v; uint4 q[2]; } u;
    const uint4* p = (const uint4*)(rowbase + k0 + kh * 16);
    u.q[0] = p[0];
    u.q[1] = p[1];
    return u.v;
}

// ---------- prep kernels ----------
__global__ void k_zero(float* __restrict__ p, long n) {
    long i = (long)blockIdx.x * blockDim.x + threadIdx.x;
    if (i < n) p[i] = 0.0f;
}
__global__ void k_cvt_x(const float* __restrict__ x, unsigned short* __restrict__ xb) {
    long i = (long)blockIdx.x * blockDim.x + threadIdx.x;
    if (i < (long)NNODE * NFEAT) xb[i] = f2bf(x[i]);
}
// W_heads [8][256][64] -> W1^T bf16 [512][256]
__global__ void k_w1t(const float* __restrict__ W, unsigned short* __restrict__ wt) {
    int i = blockIdx.x * blockDim.x + threadIdx.x;
    if (i >= NH * NFEAT) return;
    int n = i >> 8, k = i & 255;
    wt[(size_t)n * NFEAT + k] = f2bf(W[((size_t)(n >> 6) * NFEAT + k) * NHID + (n & 63)]);
}
// W_out [512][40] -> W2^T bf16 [48][512] (zero-padded cols 40..47)
__global__ void k_w2t(const float* __restrict__ W, unsigned short* __restrict__ wt) {
    int i = blockIdx.x * blockDim.x + threadIdx.x;
    if (i >= 48 * NH) return;
    int n = i >> 9, k = i & 511;
    float v = (n < NCLASS) ? W[(size_t)k * NCLASS + n] : 0.0f;
    wt[(size_t)n * NH + k] = f2bf(v);
}

// ---------- WMMA GEMM 1: Wh = leaky_relu(x @ Wcat)  [50000x256]@[256x512] ----------
// wave computes a 64(M) x 64(N) block: 16 accumulators, 16 WMMA per 8 loads per K-step.
// 128-thread blocks (1 wave/SIMD) + waves_per_eu(1): full VGPR budget, no spills.
__global__ void __launch_bounds__(128)
__attribute__((amdgpu_waves_per_eu(1)))
k_gemm1(const unsigned short* __restrict__ A,
        const unsigned short* __restrict__ BT,
        float* __restrict__ C) {
    const int MT = NNODE / 16;          // 3125 M-tiles of 16
    int lane = threadIdx.x & 31;
    int wid = blockIdx.x * (blockDim.x >> 5) + (threadIdx.x >> 5);
    int mc = wid >> 3;                  // M-chunk of 64 rows
    int nt = wid & 7;                   // 8 N-tiles of 64
    if (mc * 4 >= MT) return;           // wave-uniform guard (EXEC stays all-ones)
    int kh = lane >> 4;
    int ml = lane & 15;
    floatx8 acc[4][4];
#pragma unroll
    for (int mi = 0; mi < 4; ++mi)
#pragma unroll
        for (int ni = 0; ni < 4; ++ni) acc[mi][ni] = (floatx8){};
    const unsigned short* arow[4];
#pragma unroll
    for (int mi = 0; mi < 4; ++mi) {
        int r = mc * 64 + mi * 16 + ml;
        arow[mi] = A + (size_t)(r < NNODE ? r : NNODE - 1) * NFEAT;  // clamped
    }
    const unsigned short* brow[4];
#pragma unroll
    for (int ni = 0; ni < 4; ++ni)
        brow[ni] = BT + (size_t)(nt * 64 + ni * 16 + ml) * NFEAT;
    for (int k0 = 0; k0 < NFEAT; k0 += 32) {
        bhalf16 b[4];
#pragma unroll
        for (int ni = 0; ni < 4; ++ni) b[ni] = load_b_frag_p(brow[ni], k0, kh);
#pragma unroll
        for (int mi = 0; mi < 4; ++mi) {
            bhalf16 a = load_a_frag_p(arow[mi], k0, kh);
#pragma unroll
            for (int ni = 0; ni < 4; ++ni)
                acc[mi][ni] = __builtin_amdgcn_wmma_f32_16x16x32_bf16(
                    false, a, false, b[ni], (short)0, acc[mi][ni], false, false);
        }
    }
#pragma unroll
    for (int mi = 0; mi < 4; ++mi) {
        if (mc * 4 + mi >= MT) continue;
        int rbase = mc * 64 + mi * 16 + kh * 8;
#pragma unroll
        for (int ni = 0; ni < 4; ++ni)
#pragma unroll
            for (int r = 0; r < 8; ++r) {
                float v = acc[mi][ni][r];
                v = v > 0.0f ? v : ALPHA * v;
                C[(size_t)(rbase + r) * NH + nt * 64 + ni * 16 + ml] = v;
            }
    }
}

// ---------- WMMA GEMM 2: Wh2 = leaky_relu(h @ W_out)  [50000x512]@[512x40] ----------
// wave computes 32(M) x 48(N): 6 accumulators
__global__ void __launch_bounds__(128)
__attribute__((amdgpu_waves_per_eu(1)))
k_gemm2(const unsigned short* __restrict__ A,
        const unsigned short* __restrict__ BT,
        float* __restrict__ C) {
    const int MT = NNODE / 16;          // 3125
    int lane = threadIdx.x & 31;
    int mc = blockIdx.x * (blockDim.x >> 5) + (threadIdx.x >> 5);  // chunk of 32 rows
    if (mc * 2 >= MT) return;
    int kh = lane >> 4;
    int ml = lane & 15;
    floatx8 acc[2][3];
#pragma unroll
    for (int mi = 0; mi < 2; ++mi)
#pragma unroll
        for (int ni = 0; ni < 3; ++ni) acc[mi][ni] = (floatx8){};
    const unsigned short* arow[2];
#pragma unroll
    for (int mi = 0; mi < 2; ++mi) {
        int r = mc * 32 + mi * 16 + ml;
        arow[mi] = A + (size_t)(r < NNODE ? r : NNODE - 1) * NH;
    }
    const unsigned short* brow[3];
#pragma unroll
    for (int ni = 0; ni < 3; ++ni)
        brow[ni] = BT + (size_t)(ni * 16 + ml) * NH;
    for (int k0 = 0; k0 < NH; k0 += 32) {
        bhalf16 b[3];
#pragma unroll
        for (int ni = 0; ni < 3; ++ni) b[ni] = load_b_frag_p(brow[ni], k0, kh);
#pragma unroll
        for (int mi = 0; mi < 2; ++mi) {
            bhalf16 a = load_a_frag_p(arow[mi], k0, kh);
#pragma unroll
            for (int ni = 0; ni < 3; ++ni)
                acc[mi][ni] = __builtin_amdgcn_wmma_f32_16x16x32_bf16(
                    false, a, false, b[ni], (short)0, acc[mi][ni], false, false);
        }
    }
#pragma unroll
    for (int mi = 0; mi < 2; ++mi) {
        if (mc * 2 + mi >= MT) continue;
        int rbase = mc * 32 + mi * 16 + kh * 8;
#pragma unroll
        for (int ni = 0; ni < 3; ++ni) {
            int colb = ni * 16 + ml;
            if (colb < NCLASS) {
#pragma unroll
                for (int r = 0; r < 8; ++r) {
                    float v = acc[mi][ni][r];
                    v = v > 0.0f ? v : ALPHA * v;
                    C[(size_t)(rbase + r) * NCLASS + colb] = v;
                }
            }
        }
    }
}

// ---------- per-(node,head) attention projections e1 = Wh.a[:64], e2 = Wh.a[64:] ----------
__global__ void k_escore1(const float* __restrict__ Wh, const float* __restrict__ ah,
                          float* __restrict__ e1, float* __restrict__ e2) {
    int lane = threadIdx.x & 31;
    int wid = blockIdx.x * (blockDim.x >> 5) + (threadIdx.x >> 5);
    if (wid >= NNODE * NHEADS) return;
    int node = wid >> 3, head = wid & 7;
    float2 wh = *(const float2*)(Wh + (size_t)node * NH + head * NHID + lane * 2);
    float2 a1 = *(const float2*)(ah + (size_t)head * 2 * NHID + lane * 2);
    float2 a2 = *(const float2*)(ah + (size_t)head * 2 * NHID + NHID + lane * 2);
    float s1 = wh.x * a1.x + wh.y * a1.y;
    float s2 = wh.x * a2.x + wh.y * a2.y;
#pragma unroll
    for (int off = 16; off; off >>= 1) { s1 += __shfl_xor(s1, off); s2 += __shfl_xor(s2, off); }
    if (lane == 0) { e1[(size_t)node * 8 + head] = s1; e2[(size_t)node * 8 + head] = s2; }
}

// ---------- layer-1 edge softmax passes ----------
__global__ void k_edgeA1(const int* __restrict__ row, const int* __restrict__ col,
                         const float* __restrict__ e1, const float* __restrict__ e2,
                         float* __restrict__ sc, unsigned* __restrict__ mkey) {
    int e = blockIdx.x * blockDim.x + threadIdx.x;
    if (e >= NEDGE) return;
    int r = row[e], c = col[e];
    float4 a0 = *(const float4*)(e1 + (size_t)r * 8);
    float4 a1 = *(const float4*)(e1 + (size_t)r * 8 + 4);
    float4 b0 = *(const float4*)(e2 + (size_t)c * 8);
    float4 b1 = *(const float4*)(e2 + (size_t)c * 8 + 4);
    float s[8] = {a0.x + b0.x, a0.y + b0.y, a0.z + b0.z, a0.w + b0.w,
                  a1.x + b1.x, a1.y + b1.y, a1.z + b1.z, a1.w + b1.w};
    *(float4*)(sc + (size_t)e * 8)     = make_float4(s[0], s[1], s[2], s[3]);
    *(float4*)(sc + (size_t)e * 8 + 4) = make_float4(s[4], s[5], s[6], s[7]);
    unsigned* mk = mkey + (size_t)r * 8;
#pragma unroll
    for (int h = 0; h < 8; ++h) atomicMax(&mk[h], fenc(s[h]));
}
__global__ void k_edgeB1(const int* __restrict__ row, const unsigned* __restrict__ mkey,
                         float* __restrict__ sc, float* __restrict__ den) {
    int e = blockIdx.x * blockDim.x + threadIdx.x;
    if (e >= NEDGE) return;
    int r = row[e];
#pragma unroll
    for (int h = 0; h < 8; ++h) {
        float m = fdec(mkey[(size_t)r * 8 + h]);
        float ex = __expf(sc[(size_t)e * 8 + h] - m);
        sc[(size_t)e * 8 + h] = ex;
        atomicAdd(&den[(size_t)r * 8 + h], ex);
    }
}
// SpMM scatter: hp[row] += att * Wh[col]   (wave per edge, 512 dims coalesced)
__global__ void k_edgeC1(const int* __restrict__ row, const int* __restrict__ col,
                         const float* __restrict__ sc, const float* __restrict__ den,
                         const float* __restrict__ Wh, float* __restrict__ hp) {
    int lane = threadIdx.x & 31;
    int e = blockIdx.x * (blockDim.x >> 5) + (threadIdx.x >> 5);
    if (e >= NEDGE) return;
    int r = row[e], c = col[e];
    float4 ex0 = *(const float4*)(sc + (size_t)e * 8);
    float4 ex1 = *(const float4*)(sc + (size_t)e * 8 + 4);
    float4 d0  = *(const float4*)(den + (size_t)r * 8);
    float4 d1  = *(const float4*)(den + (size_t)r * 8 + 4);
    float att[8] = {ex0.x / d0.x, ex0.y / d0.y, ex0.z / d0.z, ex0.w / d0.w,
                    ex1.x / d1.x, ex1.y / d1.y, ex1.z / d1.z, ex1.w / d1.w};
    const float* whc = Wh + (size_t)c * NH;
    float* hpr = hp + (size_t)r * NH;
#pragma unroll
    for (int j = 0; j < 16; ++j) {      // dim>>6 == j>>1 is lane-uniform
        int dim = j * 32 + lane;
        atomicAdd(&hpr[dim], att[j >> 1] * whc[dim]);
    }
}
// elu(h') -> bf16 copy for layer 2 GEMM
__global__ void k_elu_h(const float* __restrict__ hp, unsigned short* __restrict__ hb) {
    long i = (long)blockIdx.x * blockDim.x + threadIdx.x;
    if (i >= (long)NNODE * NH) return;
    float v = hp[i];
    float y = v > 0.0f ? v : __expf(v) - 1.0f;
    hb[i] = f2bf(y);
}

// ---------- layer-2 attention + edge passes ----------
__global__ void k_escore2(const float* __restrict__ Wh2, const float* __restrict__ ao,
                          float* __restrict__ e1o, float* __restrict__ e2o) {
    int lane = threadIdx.x & 31;
    int node = blockIdx.x * (blockDim.x >> 5) + (threadIdx.x >> 5);
    if (node >= NNODE) return;
    const float* w = Wh2 + (size_t)node * NCLASS;
    float s1 = w[lane] * ao[lane];
    float s2 = w[lane] * ao[NCLASS + lane];
    if (lane < 8) {
        s1 += w[32 + lane] * ao[32 + lane];
        s2 += w[32 + lane] * ao[NCLASS + 32 + lane];
    }
#pragma unroll
    for (int off = 16; off; off >>= 1) { s1 += __shfl_xor(s1, off); s2 += __shfl_xor(s2, off); }
    if (lane == 0) { e1o[node] = s1; e2o[node] = s2; }
}
__global__ void k_edgeA2(const int* __restrict__ row, const int* __restrict__ col,
                         const float* __restrict__ e1o, const float* __restrict__ e2o,
                         float* __restrict__ sc2, unsigned* __restrict__ mkey2) {
    int e = blockIdx.x * blockDim.x + threadIdx.x;
    if (e >= NEDGE) return;
    float s = e1o[row[e]] + e2o[col[e]];
    sc2[e] = s;
    atomicMax(&mkey2[row[e]], fenc(s));
}
__global__ void k_edgeB2(const int* __restrict__ row, const unsigned* __restrict__ mkey2,
                         float* __restrict__ sc2, float* __restrict__ den2) {
    int e = blockIdx.x * blockDim.x + threadIdx.x;
    if (e >= NEDGE) return;
    int r = row[e];
    float ex = __expf(sc2[e] - fdec(mkey2[r]));
    sc2[e] = ex;
    atomicAdd(&den2[r], ex);
}
__global__ void k_edgeC2(const int* __restrict__ row, const int* __restrict__ col,
                         const float* __restrict__ sc2, const float* __restrict__ den2,
                         const float* __restrict__ Wh2, float* __restrict__ out) {
    int lane = threadIdx.x & 31;
    int e = blockIdx.x * (blockDim.x >> 5) + (threadIdx.x >> 5);
    if (e >= NEDGE) return;
    int r = row[e], c = col[e];
    float att = sc2[e] / den2[r];
    const float* w = Wh2 + (size_t)c * NCLASS;
    float* o = out + (size_t)r * NCLASS;
    atomicAdd(&o[lane], att * w[lane]);
    if (lane < 8) atomicAdd(&o[32 + lane], att * w[32 + lane]);
}
// elu + log_softmax in place on d_out (wave per node)
__global__ void k_final(float* __restrict__ out) {
    int lane = threadIdx.x & 31;
    int node = blockIdx.x * (blockDim.x >> 5) + (threadIdx.x >> 5);
    if (node >= NNODE) return;
    float* o = out + (size_t)node * NCLASS;
    float v0 = o[lane];
    float v1 = (lane < 8) ? o[32 + lane] : -__builtin_inff();
    v0 = v0 > 0.0f ? v0 : __expf(v0) - 1.0f;
    if (lane < 8) v1 = v1 > 0.0f ? v1 : __expf(v1) - 1.0f;
    float m = fmaxf(v0, v1);
#pragma unroll
    for (int off = 16; off; off >>= 1) m = fmaxf(m, __shfl_xor(m, off));
    float s = __expf(v0 - m) + ((lane < 8) ? __expf(v1 - m) : 0.0f);
#pragma unroll
    for (int off = 16; off; off >>= 1) s += __shfl_xor(s, off);
    float ls = m + __logf(s);
    o[lane] = v0 - ls;
    if (lane < 8) o[32 + lane] = v1 - ls;
}

// ---------- host launcher ----------
extern "C" void kernel_launch(void* const* d_in, const int* in_sizes, int n_in,
                              void* d_out, int out_size, void* d_ws, size_t ws_size,
                              hipStream_t stream) {
    const float* x       = (const float*)d_in[0];
    const int*   eidx    = (const int*)d_in[1];
    const float* W_heads = (const float*)d_in[2];
    const float* a_heads = (const float*)d_in[3];
    const float* W_out   = (const float*)d_in[4];
    const float* a_out   = (const float*)d_in[5];
    const int* row = eidx;
    const int* col = eidx + NEDGE;
    float* out = (float*)d_out;

    char* ws = (char*)d_ws;
    size_t off = 0;
    auto alloc = [&](size_t bytes) -> void* {
        off = (off + 255) & ~(size_t)255;
        void* p = ws + off;
        off += bytes;
        return p;
    };
    unsigned short* xb   = (unsigned short*)alloc((size_t)NNODE * NFEAT * 2);
    unsigned short* w1t  = (unsigned short*)alloc((size_t)NH * NFEAT * 2);
    unsigned short* w2t  = (unsigned short*)alloc((size_t)48 * NH * 2);
    float*          Wh   = (float*)alloc((size_t)NNODE * NH * 4);
    float*          e1   = (float*)alloc((size_t)NNODE * 8 * 4);
    float*          e2   = (float*)alloc((size_t)NNODE * 8 * 4);
    unsigned*       mkey = (unsigned*)alloc((size_t)NNODE * 8 * 4);
    float*          den  = (float*)alloc((size_t)NNODE * 8 * 4);
    float*          sc   = (float*)alloc((size_t)NEDGE * 8 * 4);
    float*          hp   = (float*)alloc((size_t)NNODE * NH * 4);
    unsigned short* hb   = (unsigned short*)alloc((size_t)NNODE * NH * 2);
    float*          Wh2  = (float*)alloc((size_t)NNODE * NCLASS * 4);
    float*          e1o  = (float*)alloc((size_t)NNODE * 4);
    float*          e2o  = (float*)alloc((size_t)NNODE * 4);
    unsigned*       mkey2= (unsigned*)alloc((size_t)NNODE * 4);
    float*          den2 = (float*)alloc((size_t)NNODE * 4);
    float*          sc2  = (float*)alloc((size_t)NEDGE * 4);
    (void)ws_size; (void)in_sizes; (void)n_in; (void)out_size;

    const int T = 256;
    auto blocks = [](long n, int t) { return (int)((n + t - 1) / t); };

    // zero accumulators (harness poisons buffers; must re-init every call)
    k_zero<<<blocks((long)NNODE * NH, T), T, 0, stream>>>(hp, (long)NNODE * NH);
    k_zero<<<blocks((long)NNODE * 8, T), T, 0, stream>>>(den, (long)NNODE * 8);
    k_zero<<<blocks((long)NNODE * 8, T), T, 0, stream>>>((float*)mkey, (long)NNODE * 8);
    k_zero<<<blocks((long)NNODE, T), T, 0, stream>>>(den2, (long)NNODE);
    k_zero<<<blocks((long)NNODE, T), T, 0, stream>>>((float*)mkey2, (long)NNODE);
    k_zero<<<blocks((long)NNODE * NCLASS, T), T, 0, stream>>>(out, (long)NNODE * NCLASS);

    // precision conversion / weight transposes
    k_cvt_x<<<blocks((long)NNODE * NFEAT, T), T, 0, stream>>>(x, xb);
    k_w1t<<<blocks((long)NH * NFEAT, T), T, 0, stream>>>(W_heads, w1t);
    k_w2t<<<blocks((long)48 * NH, T), T, 0, stream>>>(W_out, w2t);

    // layer 1: WMMA GEMM + attention softmax + SpMM
    {
        const int MT = NNODE / 16;                 // 3125
        int mchunks = (MT + 3) / 4;                // 782
        long waves = (long)mchunks * 8;            // 6256 waves
        k_gemm1<<<blocks(waves * 32, 128), 128, 0, stream>>>(xb, w1t, Wh);
    }
    k_escore1<<<blocks((long)NNODE * NHEADS * 32, T), T, 0, stream>>>(Wh, a_heads, e1, e2);
    k_edgeA1<<<blocks(NEDGE, T), T, 0, stream>>>(row, col, e1, e2, sc, mkey);
    k_edgeB1<<<blocks(NEDGE, T), T, 0, stream>>>(row, mkey, sc, den);
    k_edgeC1<<<blocks((long)NEDGE * 32, T), T, 0, stream>>>(row, col, sc, den, Wh, hp);
    k_elu_h<<<blocks((long)NNODE * NH, T), T, 0, stream>>>(hp, hb);

    // layer 2
    {
        const int MT = NNODE / 16;                 // 3125
        long waves = (MT + 1) / 2;                 // 1563 waves
        k_gemm2<<<blocks(waves * 32, 128), 128, 0, stream>>>(hb, w2t, Wh2);
    }
    k_escore2<<<blocks((long)NNODE * 32, T), T, 0, stream>>>(Wh2, a_out, e1o, e2o);
    k_edgeA2<<<blocks(NEDGE, T), T, 0, stream>>>(row, col, e1o, e2o, sc2, mkey2);
    k_edgeB2<<<blocks(NEDGE, T), T, 0, stream>>>(row, mkey2, sc2, den2);
    k_edgeC2<<<blocks((long)NEDGE * 32, T), T, 0, stream>>>(row, col, sc2, den2, Wh2, out);
    k_final<<<blocks((long)NNODE * 32, T), T, 0, stream>>>(out);
}
